// regionAgg_layer_2525440770049
// MI455X (gfx1250) — matrix operand order repaired and checked by reading, not compiled, not run on hardware
//
#include <hip/hip_runtime.h>
#include <cstdint>

#define DEVFN __device__ __forceinline__

typedef __attribute__((ext_vector_type(16))) __bf16        v16bf;
typedef __attribute__((ext_vector_type(8)))  __bf16        v8bf;
typedef __attribute__((ext_vector_type(8)))  float         v8f;
typedef __attribute__((ext_vector_type(4)))  unsigned int  v4u;
typedef __attribute__((ext_vector_type(8)))  int           v8i;
typedef __attribute__((ext_vector_type(4)))  int           v4i;

// Problem constants (reference: B,C,N,R = 8,128,65536,512)
constexpr int B_  = 8;
constexpr int C_  = 128;
constexpr int N_  = 65536;
constexpr int R_  = 512;

constexpr int KC     = 64;   // K-chunk (fp32 elements) streamed per iteration
constexpr int PITCH  = 68;   // fp32 LDS row pitch (64 + 4 pad dwords, from TDM pad feature)
constexpr int PITCHB = 72;   // bf16 plane row pitch (144B -> rows 36 banks apart, conflict-free)
constexpr int MT     = 32;   // channel tile per workgroup
constexpr int RT     = 64;   // region tile per workgroup

#if __has_builtin(__builtin_amdgcn_tensor_load_to_lds)
#define HAVE_TDM 1
#else
#define HAVE_TDM 0
#endif

#if HAVE_TDM
// 2D Tensor-DMA descriptor (D# group0 + group1, ISA ch.8); 6-arg clang-23 builtin.
// Tile = tile_rows x KC fp32, row stride = stride_elems.
// pad_enable, interval=64 dwords, amount=4 dwords -> LDS pitch 68 floats.
DEVFN void tdm_load_tile(unsigned lds_addr, const float* g,
                         unsigned tensor_rows, unsigned tile_rows,
                         unsigned stride_elems)
{
  unsigned long long ga = (unsigned long long)(uintptr_t)g;
  v4u g0;
  g0.x = 1u;                                   // count=1 (valid user descriptor)
  g0.y = lds_addr;                             // LDS byte address
  g0.z = (unsigned)ga;                         // global_addr[31:0]
  g0.w = ((unsigned)(ga >> 32) & 0x01FFFFFFu)  // global_addr[56:32]
         | (2u << 30);                         // type = 2 ("image")
  v8i g1;
  g1[0] = (int)((2u << 16)     // data_size = 4 bytes
              | (1u << 20)     // pad_enable
              | (5u << 22)     // pad_interval: 64 dwords
              | (3u << 25));   // pad_amount: 4 dwords
  g1[1] = (int)(((unsigned)N_ & 0xFFFFu) << 16);                         // tensor_dim0 lo
  g1[2] = (int)(((unsigned)N_ >> 16) | ((tensor_rows & 0xFFFFu) << 16)); // d0 hi | d1 lo
  g1[3] = (int)((tensor_rows >> 16) | ((unsigned)KC << 16));             // d1 hi | tile_dim0
  g1[4] = (int)(tile_rows & 0xFFFFu);                                    // tile_dim1, tile_dim2=0
  g1[5] = (int)stride_elems;                                             // dim0_stride[31:0]
  g1[6] = 0;
  g1[7] = 0;
  v4i z4 = {};
  v8i z8 = {};
  __builtin_amdgcn_tensor_load_to_lds(g0, g1, z4, z4, z8, 0);
}
#else
// Fallback: cooperative float4 copies into the same padded LDS layout.
DEVFN void coop_copy(float* dX, float* dM, const float* xg, const float* mg,
                     int k, int tid)
{
  for (int t = tid; t < MT * (KC / 4); t += 256) {
    int row = t / (KC / 4), q = t % (KC / 4);
    float4 v = *(const float4*)(xg + (size_t)row * N_ + k + q * 4);
    *(float4*)(dX + row * PITCH + q * 4) = v;
  }
  for (int t = tid; t < RT * (KC / 4); t += 256) {
    int row = t / (KC / 4), q = t % (KC / 4);
    float4 v = *(const float4*)(mg + (size_t)row * N_ + k + q * 4);
    *(float4*)(dM + row * PITCH + q * 4) = v;
  }
}
#endif

// Split 8 contiguous fp32 into bf16 hi/lo planes (lo = v - (float)hi is exact,
// so hi*hi + hi*lo + lo*hi recovers ~fp32 accuracy). One 16B store per plane.
DEVFN void conv_group(const float* src, __bf16* dhi, __bf16* dlo)
{
  float4 a = *(const float4*)src;
  float4 b = *(const float4*)(src + 4);
  float v[8] = {a.x, a.y, a.z, a.w, b.x, b.y, b.z, b.w};
  v8bf hi, lo;
#pragma unroll
  for (int j = 0; j < 8; ++j) {
    __bf16 h = (__bf16)v[j];
    hi[j] = h;
    lo[j] = (__bf16)(v[j] - (float)h);
  }
  *(v8bf*)dhi = hi;
  *(v8bf*)dlo = lo;
}

// A fragment (16x32 bf16): per lane needs K..K+7 and K+16..K+23 -> two v8bf.
DEVFN v16bf ld_frag_a(const __bf16* base)
{
  v8bf l = *(const v8bf*)(base);
  v8bf h = *(const v8bf*)(base + 16);
  return __builtin_shufflevector(l, h, 0,1,2,3,4,5,6,7,8,9,10,11,12,13,14,15);
}

// B fragment (32x16 bf16): per lane needs K..K+7 and K+16..K+23 -> two v8bf.
DEVFN v16bf ld_frag_b(const __bf16* base)
{
  v8bf l = *(const v8bf*)(base);
  v8bf h = *(const v8bf*)(base + 16);
  return __builtin_shufflevector(l, h, 0,1,2,3,4,5,6,7,8,9,10,11,12,13,14,15);
}

DEVFN v8f wmma_bf16(v16bf a, v16bf b, v8f c)
{
  return __builtin_amdgcn_wmma_f32_16x16x32_bf16(
      false, a, false, b, (short)0, c, false, false);
}

__global__ __launch_bounds__(256)
void regionAgg_kernel(const float* __restrict__ x,
                      const float* __restrict__ mask,
                      float* __restrict__ out)
{
  // fp32 staging (TDM target), double-buffered: 52224 B
  __shared__ float sX[2][MT * PITCH];
  __shared__ float sM[2][RT * PITCH];
  // bf16 hi/lo planes, single-buffered (publish barrier separates
  // this iteration's reads from next iteration's writes): 27648 B
  __shared__ __bf16 pXh[MT * PITCHB], pXl[MT * PITCHB];
  __shared__ __bf16 pMh[RT * PITCHB], pMl[RT * PITCHB];

  const int b  = blockIdx.z;
  const int c0 = blockIdx.y * MT;
  const int r0 = blockIdx.x * RT;

  const int tid  = threadIdx.x;
  const int lane = tid & 31;
  const int wv   = tid >> 5;      // 8 waves
  const int lr   = lane & 15;
  const int kh   = lane >> 4;     // lane half selects K sub-block

  const int m_off = (wv >> 2) << 4;  // wave's 16-row block within MT (0,16)
  const int r_off = (wv & 3) << 4;   // wave's 16-col block within RT (0..48)

  const float* xg = x    + ((size_t)b * C_ + c0) * N_;
  const float* mg = mask + ((size_t)b * R_ + r0) * N_;

  // Per-lane bf16-plane fragment bases (ISA 7.12.2 layouts).
  const int aBase = (m_off + lr) * PITCHB + kh * 8;   // A: 16x32 bf16
  const int bBase = (r_off + lr) * PITCHB + kh * 8;   // B: 32x16 bf16

  v8f acc = {};

  const int NK = N_ / KC;  // 1024 chunks
  int cur = 0;

#if HAVE_TDM
  if (wv == 0) {
    tdm_load_tile((unsigned)(uintptr_t)&sX[0][0], xg, C_, MT, N_);
    tdm_load_tile((unsigned)(uintptr_t)&sM[0][0], mg, R_, RT, N_);
    __builtin_amdgcn_s_wait_tensorcnt(0);
  }
  __syncthreads();
#else
  coop_copy(&sX[0][0], &sM[0][0], xg, mg, 0, tid);
  __syncthreads();
#endif

  for (int i = 0; i < NK; ++i) {
    const int nxt = cur ^ 1;

#if HAVE_TDM
    if (wv == 0 && i + 1 < NK) {
      const int k = (i + 1) * KC;
      tdm_load_tile((unsigned)(uintptr_t)&sX[nxt][0], xg + k, C_, MT, N_);
      tdm_load_tile((unsigned)(uintptr_t)&sM[nxt][0], mg + k, R_, RT, N_);
    }
#endif

    // ---- cooperative fp32 -> bf16 hi/lo conversion (once per workgroup) ----
    // 96 rows (32 X + 64 M) x 8 groups of 8 floats = 768 groups, 3 per thread.
    {
      const float* sx = &sX[cur][0];
      const float* sm = &sM[cur][0];
#pragma unroll
      for (int gi = 0; gi < 3; ++gi) {
        const int g   = tid + gi * 256;
        const int row = g >> 3;
        const int c8  = (g & 7) * 8;
        if (row < MT) {
          conv_group(sx + row * PITCH + c8,
                     &pXh[row * PITCHB + c8], &pXl[row * PITCHB + c8]);
        } else {
          const int r = row - MT;
          conv_group(sm + r * PITCH + c8,
                     &pMh[r * PITCHB + c8], &pMl[r * PITCHB + c8]);
        }
      }
    }
    __syncthreads();  // planes ready

    // ---- WMMA phase: pure ds_load fragment assembly, no conversion VALU ----
#pragma unroll
    for (int kk = 0; kk < KC; kk += 32) {
      v16bf ah = ld_frag_a(&pXh[aBase + kk]);
      v16bf al = ld_frag_a(&pXl[aBase + kk]);
      v16bf bh = ld_frag_b(&pMh[bBase + kk]);
      v16bf bl = ld_frag_b(&pMl[bBase + kk]);
      acc = wmma_bf16(ah, bh, acc);
      acc = wmma_bf16(ah, bl, acc);
      acc = wmma_bf16(al, bh, acc);
    }

#if HAVE_TDM
    if (i + 1 < NK) {
      if (wv == 0) __builtin_amdgcn_s_wait_tensorcnt(0);
      __syncthreads();  // publish fp32[nxt]; also fences plane reuse
    } else {
      __syncthreads();
    }
#else
    __syncthreads();
    if (i + 1 < NK) {
      coop_copy(&sX[nxt][0], &sM[nxt][0], xg, mg, (i + 1) * KC, tid);
      __syncthreads();
    }
#endif
    cur = nxt;
  }

  // C/D layout: VGPR e -> M = e + 8*(lane>=16), N = lane & 15
  const int orow0 = c0 + m_off + kh * 8;
  const int ocol  = r0 + r_off + lr;
  float* op = out + ((size_t)b * C_ + orow0) * R_ + ocol;
#pragma unroll
  for (int e = 0; e < 8; ++e)
    op[(size_t)e * R_] = acc[e];
}

extern "C" void kernel_launch(void* const* d_in, const int* in_sizes, int n_in,
                              void* d_out, int out_size, void* d_ws, size_t ws_size,
                              hipStream_t stream)
{
  (void)in_sizes; (void)n_in; (void)out_size; (void)d_ws; (void)ws_size;
  const float* x    = (const float*)d_in[0];
  const float* mask = (const float*)d_in[1];
  float* out        = (float*)d_out;

  dim3 grid(R_ / RT, C_ / MT, B_);  // 8 x 4 x 8 = 256 workgroups
  hipLaunchKernelGGL(regionAgg_kernel, grid, dim3(256), 0, stream, x, mask, out);
}
